// SMN_56891136803576
// MI455X (gfx1250) — matrix operand order repaired and checked by reading, not compiled, not run on hardware
//
#include <hip/hip_runtime.h>
#include <cmath>

// ---------------------------------------------------------------------------
// Types for CDNA5 WMMA (wave32, V_WMMA_F32_16X16X32_BF16)
// ---------------------------------------------------------------------------
typedef __bf16 bf16_t;
typedef __attribute__((ext_vector_type(16))) __bf16 v16bf;
typedef __attribute__((ext_vector_type(8)))  __bf16 v8bf;
typedef __attribute__((ext_vector_type(8)))  float  v8f;

__device__ __forceinline__ v16bf smn_cat16(v8bf lo, v8bf hi) {
  return __builtin_shufflevector(lo, hi, 0,1,2,3,4,5,6,7,8,9,10,11,12,13,14,15);
}

// A-matrix 16x32 bf16 fragment: lanes 0-15 => M=lane, K range
// [half*16 .. half*16+15] where half = lane>>4.
__device__ __forceinline__ v16bf smn_load_a(const bf16_t* __restrict__ A,
                                            long long lda, int row, int k0, int half) {
  const bf16_t* p = A + (long long)row * lda + k0 + half * 16;
  v8bf lo = *(const v8bf*)(p);
  v8bf hi = *(const v8bf*)(p + 8);
  return smn_cat16(lo, hi);
}

// B-matrix 32x16 bf16 fragment from global: lane col = lane&15,
// K range = (lane>>4)*16 .. +15, B stored row-major [N, Kp].
__device__ __forceinline__ v16bf smn_load_b(const bf16_t* __restrict__ B,
                                            long long ldb, int col, int k0, int half) {
  const bf16_t* p = B + (long long)col * ldb + k0 + half * 16;
  v8bf lo = *(const v8bf*)(p);
  v8bf hi = *(const v8bf*)(p + 8);
  return smn_cat16(lo, hi);
}

__device__ __forceinline__ v8f smn_wmma(v16bf a, v16bf b, v8f c) {
  return __builtin_amdgcn_wmma_f32_16x16x32_bf16(false, a, false, b, (short)0, c,
                                                 false, false);
}

// Async global->LDS copy of one 16B chunk (ASYNCcnt path). %0 = LDS byte
// offset within the workgroup allocation, %1 = 64-bit global source address.
// NOTE: no "m" constraints -- AMDGPU ISel cannot match LDS memory operands in
// inline asm; visibility of the LDS write is ensured by the dummy padding
// stores in the kernel (see below).
__device__ __forceinline__ void smn_async_b128(unsigned ldsOff, const void* gptr) {
  asm volatile("global_load_async_to_lds_b128 %0, %1, off"
               :
               : "v"(ldsOff), "v"((unsigned long long)(uintptr_t)gptr)
               : "memory");
}
__device__ __forceinline__ void smn_wait_async0() {
  asm volatile("s_wait_asynccnt 0x0" ::: "memory");
}

// ---------------------------------------------------------------------------
// LDS-staged GEMM: C[M,N] (f32, +bias) = A[M,Kp](bf16) @ B[N,Kp](bf16)^T
// block = 256 threads = 8 waves = 4 M-tiles x 2 N-subtiles -> 64x64 per block.
// B tile (64 cols x 32 k) staged via async global->LDS loads (double buffered,
// padded stride 40 halves so the 16 ds_load_b128 lanes hit distinct bank
// groups). Requires M % 64 == 0, N % 64 == 0, Kp % 32 == 0.
// ---------------------------------------------------------------------------
#define BSTR 40
__global__ void smn_gemm_f32_lds(const bf16_t* __restrict__ A, long long lda,
                                 const bf16_t* __restrict__ B, long long ldb,
                                 float* __restrict__ C, long long ldc,
                                 int M, int N, int Kp,
                                 const float* __restrict__ bias, int nbias) {
  __shared__ bf16_t Bs[2][64 * BSTR];     // only shared object -> LDS offset 0
  int tid  = threadIdx.x;
  int wave = tid >> 5;
  int lane = tid & 31;
  int mTile = wave >> 1;                  // 0..3
  int nSub  = wave & 1;                   // 0..1
  int m0   = blockIdx.x * 64 + mTile * 16;
  int nBlk = blockIdx.y * 64;
  int half = lane >> 4;
  int l15  = lane & 15;
  int row  = m0 + l15;

  // Staging map: thread t copies one 16B chunk: col = t>>2, k-off = (t&3)*8.
  int sCol = tid >> 2;
  int sK   = (tid & 3) * 8;
  const bf16_t* Bg = B + (long long)(nBlk + sCol) * ldb + sK;
  unsigned ldsOff0 = (unsigned)((char*)&Bs[0][sCol * BSTR + sK] - (char*)&Bs[0][0]);
  unsigned ldsOff1 = (unsigned)((char*)&Bs[1][sCol * BSTR + sK] - (char*)&Bs[0][0]);

  // Compiler-visible stores into the (never-read) pad halves 32..39 of each
  // staging row: with real runtime-offset stores to Bs in the IR, the later
  // fragment loads cannot be folded to undef even though the bulk data
  // arrives via the inline-asm async copies.
  Bs[0][sCol * BSTR + 36] = (bf16_t)0.f;
  Bs[1][sCol * BSTR + 36] = (bf16_t)0.f;

  smn_async_b128(ldsOff0, Bg);            // preload k0 = 0 into buffer 0
  smn_wait_async0();
  __syncthreads();

  v8f c0 = {}, c1 = {};
  int buf = 0;
  for (int k0 = 0; k0 < Kp; k0 += 32) {
    if (k0 + 32 < Kp) {                   // async-prefetch next B tile
      smn_async_b128(buf ? ldsOff0 : ldsOff1, Bg + k0 + 32);
      __builtin_prefetch(A + (long long)row * lda + k0 + 32, 0, 1);
    }
    v16bf a = smn_load_a(A, lda, row, k0, half);
    const bf16_t* bp0 = &Bs[buf][(nSub * 32 + l15) * BSTR + half * 16];
    const bf16_t* bp1 = bp0 + 16 * BSTR;
    v16bf b0 = smn_cat16(*(const v8bf*)bp0, *(const v8bf*)(bp0 + 8));
    v16bf b1 = smn_cat16(*(const v8bf*)bp1, *(const v8bf*)(bp1 + 8));
    c0 = smn_wmma(a, b0, c0);
    c1 = smn_wmma(a, b1, c1);
    buf ^= 1;
    smn_wait_async0();                    // next buffer resident before barrier
    __syncthreads();
  }

  int col0 = nBlk + nSub * 32 + l15, col1 = col0 + 16;
  float bv0 = (bias && col0 < nbias) ? bias[col0] : 0.f;
  float bv1 = (bias && col1 < nbias) ? bias[col1] : 0.f;
#pragma unroll
  for (int j = 0; j < 8; ++j) {
    int r = m0 + j + 8 * half;            // C/D layout: VGPR j, lane-half adds M+8
    C[(long long)r * ldc + col0] = c0[j] + bv0;
    C[(long long)r * ldc + col1] = c1[j] + bv1;
  }
}

// Simple 16x32-per-wave GEMM, bf16 output (ug @ Amat^T staging; N=224).
__global__ void smn_gemm_bf16(const bf16_t* __restrict__ A, long long lda,
                              const bf16_t* __restrict__ B, long long ldb,
                              bf16_t* __restrict__ C, long long ldc,
                              int M, int N, int Kp) {
  int wave = threadIdx.x >> 5;
  int lane = threadIdx.x & 31;
  int m0 = (blockIdx.x * 4 + wave) * 16;
  int n0 = blockIdx.y * 32;
  if (m0 >= M) return;                    // wave-uniform; EXEC stays all-ones
  int half = lane >> 4;
  int l15  = lane & 15;
  int row  = m0 + l15;
  v8f c0 = {}, c1 = {};
  for (int k0 = 0; k0 < Kp; k0 += 32) {
    v16bf a  = smn_load_a(A, lda, row, k0, half);
    v16bf b0 = smn_load_b(B, ldb, n0 + l15, k0, half);
    v16bf b1 = smn_load_b(B, ldb, n0 + 16 + l15, k0, half);
    c0 = smn_wmma(a, b0, c0);
    c1 = smn_wmma(a, b1, c1);
  }
  int col0 = n0 + l15, col1 = col0 + 16;
#pragma unroll
  for (int j = 0; j < 8; ++j) {
    int r = m0 + j + 8 * half;
    C[(long long)r * ldc + col0] = (bf16_t)c0[j];
    C[(long long)r * ldc + col1] = (bf16_t)c1[j];
  }
}

// Batched masked GEMM for m1/m2: per batch z, A[64,Kp] @ B[64,Kp]^T, valid
// 50x50 stored to conv plane OUT + z*bsO + chOff. block=128 (4 M-tiles),
// grid=(1,2,nBatch); B batch index = z / bdiv.
__global__ void smn_gemm_batched(const bf16_t* __restrict__ A, long long bsA, long long lda,
                                 const bf16_t* __restrict__ B, int bdiv, long long bsB,
                                 long long ldb,
                                 float* __restrict__ OUT, long long bsO, long long chOff,
                                 int Kp, int validM, int validN, int ldo) {
  int wave = threadIdx.x >> 5;
  int lane = threadIdx.x & 31;
  int z  = blockIdx.z;
  int m0 = wave * 16;
  int n0 = blockIdx.y * 32;
  const bf16_t* Ab = A + (long long)z * bsA;
  const bf16_t* Bb = B + (long long)(z / bdiv) * bsB;
  int half = lane >> 4;
  int l15  = lane & 15;
  int row  = m0 + l15;
  v8f c0 = {}, c1 = {};
  for (int k0 = 0; k0 < Kp; k0 += 32) {
    v16bf a  = smn_load_a(Ab, lda, row, k0, half);
    v16bf b0 = smn_load_b(Bb, ldb, n0 + l15, k0, half);
    v16bf b1 = smn_load_b(Bb, ldb, n0 + 16 + l15, k0, half);
    c0 = smn_wmma(a, b0, c0);
    c1 = smn_wmma(a, b1, c1);
  }
  float* Ob = OUT + (long long)z * bsO + chOff;
  int col0 = n0 + l15, col1 = col0 + 16;
#pragma unroll
  for (int j = 0; j < 8; ++j) {
    int r = m0 + j + 8 * half;
    if (r < validM) {
      if (col0 < validN) Ob[(long long)r * ldo + col0] = c0[j];
      if (col1 < validN) Ob[(long long)r * ldo + col1] = c1[j];
    }
  }
}

// ---------------------------------------------------------------------------
// Support kernels
// ---------------------------------------------------------------------------
__global__ void smn_zero_u4(uint4* p, long long n16) {
  long long i = blockIdx.x * (long long)blockDim.x + threadIdx.x;
  if (i < n16) p[i] = make_uint4(0u, 0u, 0u, 0u);
}

__global__ void smn_pad_w(const float* __restrict__ src, bf16_t* __restrict__ dst,
                          int sR, int sC, int dR, int dC, int transpose) {
  long long i = blockIdx.x * (long long)blockDim.x + threadIdx.x;
  if (i >= (long long)dR * dC) return;
  int r = (int)(i / dC), c = (int)(i % dC);
  float v = 0.f;
  if (transpose) { if (c < sR && r < sC) v = src[(long long)c * sC + r]; }
  else           { if (r < sR && c < sC) v = src[(long long)r * sC + c]; }
  dst[i] = (bf16_t)v;
}

__global__ void smn_gather(const int* __restrict__ tok, const float* __restrict__ emb,
                           bf16_t* __restrict__ dst, long long nRows,
                           int Lv, int Lp, int Ev, int Ep) {
  long long i = blockIdx.x * (long long)blockDim.x + threadIdx.x;
  long long total = nRows * Lp * Ep;
  if (i >= total) return;
  int c = (int)(i % Ep);
  long long t = i / Ep;
  int l = (int)(t % Lp);
  long long n = t / Lp;
  float v = 0.f;
  if (l < Lv && c < Ev) {
    int id = tok[n * Lv + l];
    v = emb[(long long)id * Ev + c];
  }
  dst[i] = (bf16_t)v;
}

// GRU gates (PyTorch order r,z,n). xp/gh: [M, ldg] f32 with biases already added.
__global__ void smn_gru_gates(const float* __restrict__ xp, const float* __restrict__ gh,
                              bf16_t* __restrict__ h, bf16_t* __restrict__ ys,
                              int M, int Hv, int ldg, int ldh, int Lp, int t) {
  long long i = blockIdx.x * (long long)blockDim.x + threadIdx.x;
  if (i >= (long long)M * Hv) return;
  int j = (int)(i % Hv);
  long long m = i / Hv;
  const float* x = xp + m * ldg;
  const float* g = gh + m * ldg;
  float r = 1.f / (1.f + expf(-(x[j] + g[j])));
  float z = 1.f / (1.f + expf(-(x[Hv + j] + g[Hv + j])));
  float n = tanhf(x[2 * Hv + j] + r * g[2 * Hv + j]);
  float hp = (float)h[m * ldh + j];
  float hn = (1.f - z) * n + z * hp;
  h[m * ldh + j] = (bf16_t)hn;
  if (ys) ys[(m * Lp + t) * (long long)ldh + j] = (bf16_t)hn;
}

// Fused conv(3x3 VALID, 2->8) + ReLU + maxpool(3x3,s3): [N,2,50,50]->[N,8,16,16]
__global__ void smn_conv_pool(const float* __restrict__ x, const float* __restrict__ w,
                              const float* __restrict__ bias, float* __restrict__ out,
                              long long Ntot) {
  long long i = blockIdx.x * (long long)blockDim.x + threadIdx.x;
  long long total = Ntot * 2048;
  if (i >= total) return;
  long long n = i / 2048;
  int rem = (int)(i % 2048);
  int c  = rem >> 8;
  int p  = rem & 255;
  int py = p >> 4, px = p & 15;
  const float* x0 = x + n * 5000;
  const float* x1 = x0 + 2500;
  const float* wc = w + c * 18;
  float mx = -__builtin_inff();
#pragma unroll
  for (int dy = 0; dy < 3; ++dy) {
    int y = py * 3 + dy;
#pragma unroll
    for (int dx = 0; dx < 3; ++dx) {
      int xx = px * 3 + dx;
      float s = bias[c];
#pragma unroll
      for (int ky = 0; ky < 3; ++ky)
#pragma unroll
        for (int kx = 0; kx < 3; ++kx) {
          s += x0[(y + ky) * 50 + xx + kx] * wc[ky * 3 + kx];
          s += x1[(y + ky) * 50 + xx + kx] * wc[9 + ky * 3 + kx];
        }
      s  = fmaxf(s, 0.f);
      mx = fmaxf(mx, s);
    }
  }
  out[i] = mx;
}

__global__ void smn_f32_to_bf16(const float* __restrict__ src, bf16_t* __restrict__ dst,
                                long long n) {
  long long i = blockIdx.x * (long long)blockDim.x + threadIdx.x;
  if (i < n) dst[i] = (bf16_t)src[i];
}

__global__ void smn_tanh_bf16(const float* __restrict__ src, bf16_t* __restrict__ dst,
                              long long n) {
  long long i = blockIdx.x * (long long)blockDim.x + threadIdx.x;
  if (i < n) dst[i] = (bf16_t)tanhf(src[i]);
}

__global__ void smn_logits(const bf16_t* __restrict__ h, const float* __restrict__ w,
                           const float* __restrict__ b, float* __restrict__ out,
                           int Bn, int Hv, int ldh) {
  int i = blockIdx.x * blockDim.x + threadIdx.x;
  if (i >= Bn) return;
  float s = b[0];
  const bf16_t* hr = h + (long long)i * ldh;
  for (int j = 0; j < Hv; ++j) s += (float)hr[j] * w[j];
  out[i] = 1.f / (1.f + expf(-s));
}

// ---------------------------------------------------------------------------
// Host orchestration
// ---------------------------------------------------------------------------
extern "C" void kernel_launch(void* const* d_in, const int* in_sizes, int n_in,
                              void* d_out, int out_size, void* d_ws, size_t ws_size,
                              hipStream_t stream) {
  (void)in_sizes; (void)n_in; (void)out_size;
  const int*   utter  = (const int*)d_in[0];    // [512,10,50]
  const int*   resp   = (const int*)d_in[1];    // [512,50]
  const float* emb    = (const float*)d_in[2];  // [50000,200]
  const float* W_ih_u = (const float*)d_in[3];
  const float* W_hh_u = (const float*)d_in[4];
  const float* b_ih_u = (const float*)d_in[5];
  const float* b_hh_u = (const float*)d_in[6];
  const float* W_ih_r = (const float*)d_in[7];
  const float* W_hh_r = (const float*)d_in[8];
  const float* b_ih_r = (const float*)d_in[9];
  const float* b_hh_r = (const float*)d_in[10];
  const float* conv_w = (const float*)d_in[11];
  const float* conv_b = (const float*)d_in[12];
  const float* lin_w  = (const float*)d_in[13];
  const float* lin_b  = (const float*)d_in[14];
  const float* Amat   = (const float*)d_in[15];
  const float* W_ih_f = (const float*)d_in[16];
  const float* W_hh_f = (const float*)d_in[17];
  const float* b_ih_f = (const float*)d_in[18];
  const float* b_hh_f = (const float*)d_in[19];
  const float* flin_w = (const float*)d_in[20];
  const float* flin_b = (const float*)d_in[21];
  float* out = (float*)d_out;

  const long long Bn = 512, Uu = 10, Lv = 50, Ev = 200, Hv = 200;
  const int Lp = 64, KP = 224, G = 640;   // padded L, padded E/H, padded 3H (64-mult)
  const long long NBU = Bn * Uu;          // 5120

  // ---- scratch layout (256B aligned) ----
  char* base = (char*)d_ws;
  size_t off = 0;
  auto alloc = [&](size_t bytes) -> char* {
    char* p = base + off;
    off += (bytes + 255) & ~(size_t)255;
    return p;
  };
  bf16_t* ue   = (bf16_t*)alloc(NBU * Lp * KP * 2);  // aliased by ugA later
  bf16_t* re   = (bf16_t*)alloc(Bn * Lp * KP * 2);
  bf16_t* ug   = (bf16_t*)alloc(NBU * Lp * KP * 2);
  bf16_t* rg   = (bf16_t*)alloc(Bn * Lp * KP * 2);
  bf16_t* h_u  = (bf16_t*)alloc(NBU * KP * 2);
  bf16_t* h_r  = (bf16_t*)alloc(Bn * KP * 2);
  bf16_t* h_f  = (bf16_t*)alloc(Bn * KP * 2);
  bf16_t* Wihu = (bf16_t*)alloc((size_t)G * KP * 2);
  bf16_t* Whhu = (bf16_t*)alloc((size_t)G * KP * 2);
  bf16_t* Wihr = (bf16_t*)alloc((size_t)G * KP * 2);
  bf16_t* Whhr = (bf16_t*)alloc((size_t)G * KP * 2);
  bf16_t* AmT  = (bf16_t*)alloc((size_t)KP * KP * 2);
  bf16_t* Wlin = (bf16_t*)alloc((size_t)64 * 2048 * 2);
  bf16_t* Wihf = (bf16_t*)alloc((size_t)G * 64 * 2);
  bf16_t* Whhf = (bf16_t*)alloc((size_t)G * KP * 2);
  bf16_t* mv   = (bf16_t*)alloc(NBU * 64 * 2);
  bf16_t* plbf = (bf16_t*)alloc(NBU * 2048 * 2);
  float*  xp_u = (float*)alloc(NBU * G * 4);
  float*  gh_u = (float*)alloc(NBU * G * 4);
  float*  xp_s = (float*)alloc(Bn * G * 4);
  float*  gh_s = (float*)alloc(Bn * G * 4);
  float*  cvin = (float*)alloc(NBU * 5000 * 4);
  float*  plf  = (float*)alloc(NBU * 2048 * 4);
  float*  lino = (float*)alloc(NBU * 64 * 4);
  bf16_t* ugA  = ue;                      // ue dead after m1; safe alias
  if (off > ws_size) return;              // scratch too small: deterministic no-op

  auto gemm = [&](const bf16_t* A, long long lda, const bf16_t* B, long long ldb,
                  float* C, long long ldc, int M, int N, int Kp,
                  const float* bias, int nbias) {
    dim3 g((unsigned)(M / 64), (unsigned)(N / 64), 1);
    smn_gemm_f32_lds<<<g, 256, 0, stream>>>(A, lda, B, ldb, C, ldc, M, N, Kp,
                                            bias, nbias);
  };
  auto zero = [&](void* p, size_t bytes) {
    long long n16 = (long long)(bytes / 16);
    smn_zero_u4<<<dim3((unsigned)((n16 + 255) / 256)), 256, 0, stream>>>((uint4*)p, n16);
  };
  auto pad = [&](const float* s, bf16_t* d, int sR, int sC, int dR, int dC, int tr) {
    long long tot = (long long)dR * dC;
    smn_pad_w<<<dim3((unsigned)((tot + 255) / 256)), 256, 0, stream>>>(s, d, sR, sC,
                                                                       dR, dC, tr);
  };

  // 1) Stage weights (bf16, zero padded; N padded to 640)
  pad(W_ih_u, Wihu, 600, 200, G, KP, 0);
  pad(W_hh_u, Whhu, 600, 200, G, KP, 0);
  pad(W_ih_r, Wihr, 600, 200, G, KP, 0);
  pad(W_hh_r, Whhr, 600, 200, G, KP, 0);
  pad(Amat,   AmT,  200, 200, KP, KP, 1);  // AmT[k][h] = Amat[h][k]
  pad(lin_w,  Wlin, 50, 2048, 64, 2048, 0);
  pad(W_ih_f, Wihf, 600, 50,  G, 64, 0);
  pad(W_hh_f, Whhf, 600, 200, G, KP, 0);

  // 2) Embedding gathers (padded bf16)
  {
    long long tot = NBU * Lp * KP;
    smn_gather<<<dim3((unsigned)((tot + 255) / 256)), 256, 0, stream>>>(
        utter, emb, ue, NBU, (int)Lv, Lp, (int)Ev, KP);
    tot = Bn * Lp * KP;
    smn_gather<<<dim3((unsigned)((tot + 255) / 256)), 256, 0, stream>>>(
        resp, emb, re, Bn, (int)Lv, Lp, (int)Ev, KP);
  }

  // 3) Zero-init GRU state buffers (covers K-padding of ug/rg)
  zero(ug, NBU * Lp * KP * 2);
  zero(rg, Bn * Lp * KP * 2);
  zero(h_u, NBU * KP * 2);
  zero(h_r, Bn * KP * 2);
  zero(h_f, Bn * KP * 2);

  const long long ldSeq = (long long)Lp * KP;   // stride between sequences
  auto gates = [&](const float* xp, const float* gh, bf16_t* h, bf16_t* ys,
                   int M, int t) {
    long long tot = (long long)M * Hv;
    smn_gru_gates<<<dim3((unsigned)((tot + 255) / 256)), 256, 0, stream>>>(
        xp, gh, h, ys, M, (int)Hv, G, KP, Lp, t);
  };

  // 4) Response GRU: 512 seqs x 50 steps -> rg
  for (int t = 0; t < (int)Lv; ++t) {
    gemm(re + t * KP, ldSeq, Wihr, KP, xp_s, G, (int)Bn, G, KP, b_ih_r, 600);
    gemm(h_r, KP, Whhr, KP, gh_s, G, (int)Bn, G, KP, b_hh_r, 600);
    gates(xp_s, gh_s, h_r, rg, (int)Bn, t);
  }

  // 5) Utterance GRU: 5120 seqs x 50 steps -> ug  (must precede ugA alias!)
  for (int t = 0; t < (int)Lv; ++t) {
    gemm(ue + t * KP, ldSeq, Wihu, KP, xp_u, G, (int)NBU, G, KP, b_ih_u, 600);
    gemm(h_u, KP, Whhu, KP, gh_u, G, (int)NBU, G, KP, b_hh_u, 600);
    gates(xp_u, gh_u, h_u, ug, (int)NBU, t);
  }

  // 6) m1 = ue @ re^T  -> conv channel 0
  smn_gemm_batched<<<dim3(1, 2, (unsigned)NBU), 128, 0, stream>>>(
      ue, ldSeq, KP, re, (int)Uu, ldSeq, KP, cvin, 5000, 0, KP, 50, 50, 50);

  // 7) ugA = ug @ Amat (via AmT rows) -> bf16, overwrites ue region
  smn_gemm_bf16<<<dim3((unsigned)((NBU * Lp) / 64), KP / 32), 128, 0, stream>>>(
      ug, KP, AmT, KP, ugA, KP, (int)(NBU * Lp), KP, KP);

  // 8) m2 = ugA @ rg^T -> conv channel 1
  smn_gemm_batched<<<dim3(1, 2, (unsigned)NBU), 128, 0, stream>>>(
      ugA, ldSeq, KP, rg, (int)Uu, ldSeq, KP, cvin, 5000, 2500, KP, 50, 50, 50);

  // 9) conv + relu + maxpool -> [5120, 2048] f32, then bf16
  {
    long long tot = NBU * 2048;
    smn_conv_pool<<<dim3((unsigned)((tot + 255) / 256)), 256, 0, stream>>>(
        cvin, conv_w, conv_b, plf, NBU);
    smn_f32_to_bf16<<<dim3((unsigned)((tot + 255) / 256)), 256, 0, stream>>>(plf, plbf,
                                                                             tot);
  }

  // 10) linear 2048->50 (+bias) then tanh -> mv bf16 [512,10,64]
  gemm(plbf, 2048, Wlin, 2048, lino, 64, (int)NBU, 64, 2048, lin_b, 50);
  {
    long long tot = NBU * 64;
    smn_tanh_bf16<<<dim3((unsigned)((tot + 255) / 256)), 256, 0, stream>>>(lino, mv, tot);
  }

  // 11) Final GRU over U=10 steps on mv (K padded 50->64)
  for (int u = 0; u < (int)Uu; ++u) {
    gemm(mv + u * 64, (long long)Uu * 64, Wihf, 64, xp_s, G, (int)Bn, G, 64, b_ih_f, 600);
    gemm(h_f, KP, Whhf, KP, gh_s, G, (int)Bn, G, KP, b_hh_f, 600);
    gates(xp_s, gh_s, h_f, nullptr, (int)Bn, u);
  }

  // 12) logits + sigmoid -> d_out [512]
  smn_logits<<<dim3(2), 256, 0, stream>>>(h_f, flin_w, flin_b, out, (int)Bn, (int)Hv, KP);
}
